// Decoder_755914244448
// MI455X (gfx1250) — matrix operand order repaired and checked by reading, not compiled, not run on hardware
//
#include <hip/hip_runtime.h>
#include <cstdint>
#include <cstddef>

// ---------------- problem constants ----------------
#define BATCH   8192
#define TSTEPS  48
#define FDIM    64
#define HDIM    128
#define NGATE   512            // 4*H
#define KT_N    7              // K tiles of 32 -> K padded to 224
#define ROWS    32             // batch rows per workgroup
#define XSTRIDE 232            // u16 elems per sX row (>=224, bank-staggered)

// K layout inside x / packed weights:
//   [0..63] features, [64] prev_out, [65] const 1.0 (bias row), [66..193] h, [194..223] zero

typedef __attribute__((ext_vector_type(16))) __bf16 v16bf;
typedef __attribute__((ext_vector_type(8)))  float  v8f;

union ABOp {
    unsigned int u[8];
    uint4        u4[2];
    v16bf        v;
};
union ACCU {
    v8f    v;
    float4 f4[2];
};

__device__ __forceinline__ unsigned short f2bf(float f) {
    unsigned int u = __float_as_uint(f);
    u += 0x7FFFu + ((u >> 16) & 1u);        // round-to-nearest-even
    return (unsigned short)(u >> 16);
}
__device__ __forceinline__ unsigned int packbf(float a, float b) {
    return (unsigned int)f2bf(a) | ((unsigned int)f2bf(b) << 16);
}

#if __has_builtin(__builtin_amdgcn_rcpf)
__device__ __forceinline__ float fast_rcp(float x) { return __builtin_amdgcn_rcpf(x); }
#else
__device__ __forceinline__ float fast_rcp(float x) { return 1.0f / x; }
#endif
#if __has_builtin(__builtin_amdgcn_tanhf)
__device__ __forceinline__ float fast_tanh(float x) { return __builtin_amdgcn_tanhf(x); }
#else
__device__ __forceinline__ float fast_tanh(float x) { return tanhf(x); }
#endif
__device__ __forceinline__ float sigmoidf_(float x) {
    return fast_rcp(1.0f + __expf(-x));
}

// weight source for packed K index k, column n (Keras gate order i|f|c|o)
__device__ __forceinline__ float wsrc(const float* __restrict__ Wx,
                                      const float* __restrict__ Wh,
                                      const float* __restrict__ bias,
                                      int k, int n) {
    if (k < FDIM)                 return Wx[(size_t)(k + 1) * NGATE + n]; // feature rows 1..64
    if (k == FDIM)                return Wx[n];                           // prev_out row 0
    if (k == FDIM + 1)            return bias[n];                         // bias row (x=1.0)
    if (k >= 66 && k < 66 + HDIM) return Wh[(size_t)(k - 66) * NGATE + n];
    return 0.0f;                                                          // k >= 194
}

__global__ __launch_bounds__(256, 1)
void lstm_decoder_wmma(const float* __restrict__ feats,     // [B,T,F]
                       const float* __restrict__ init_out,  // [B,1]
                       const float* __restrict__ h0,        // [B,H]
                       const float* __restrict__ c0,        // [B,H]
                       const float* __restrict__ Wx,        // [65,512]
                       const float* __restrict__ Wh,        // [128,512]
                       const float* __restrict__ bias,      // [512]
                       const float* __restrict__ Wd,        // [128,1]
                       const float* __restrict__ bd,        // [1]
                       float* __restrict__ out)             // [B,T,1]
{
    // ---- ~310KB LDS (CDNA5 320KB WGP): weights persist for all 48 steps ----
    __shared__ unsigned int   sW[KT_N * 32 * 32 * 8];  // 229376 B, WMMA-B-packed bf16
    __shared__ unsigned short sX[ROWS * XSTRIDE];      //  14848 B, x rows in bf16
    __shared__ float          sZc[8 * 8 * 32 * 8];     //  65536 B, z in C-register layout
    __shared__ float          sWd[HDIM];               //    512 B

    const int tid  = threadIdx.x;
    const int lane = tid & 31;
    const int wv   = tid >> 5;
    const int g    = wv & 3;          // gate (i,f,c,o)
    const int mt   = wv >> 2;         // 16-row tile within the 32-row block
    const int hlf  = lane >> 4;
    const int row0 = blockIdx.x * ROWS;

    // ---------- one-time: pack weights (+bias row) into WMMA B tile layout ----------
    // sW[((kt*32 + ntile)*32 + lane)*8 + v]  <- bf16 pair K = kt*32 + 2v + 16*(lane>=16),
    // column N = ntile*16 + (lane&15).
    for (int idx = tid; idx < KT_N * 32 * 32 * 8; idx += 256) {
        const int v    = idx & 7;
        const int ln   = (idx >> 3) & 31;
        const int tile = idx >> 8;
        const int nt   = tile & 31;
        const int kt   = tile >> 5;
        const int n    = nt * 16 + (ln & 15);
        const int k0   = kt * 32 + (v << 1) + ((ln >> 4) << 4);
        sW[idx] = packbf(wsrc(Wx, Wh, bias, k0, n), wsrc(Wx, Wh, bias, k0 + 1, n));
    }
    if (tid < HDIM) sWd[tid] = Wd[tid];

    // ---------- one-time: x staging init ----------
    for (int idx = tid; idx < ROWS * XSTRIDE; idx += 256) sX[idx] = 0;
    __syncthreads();
    for (int idx = tid; idx < ROWS * HDIM; idx += 256) {
        const int m = idx >> 7, n = idx & (HDIM - 1);
        sX[m * XSTRIDE + 66 + n] = f2bf(h0[(size_t)(row0 + m) * HDIM + n]);
    }
    if (tid < ROWS) {
        sX[tid * XSTRIDE + 64] = f2bf(init_out[row0 + tid]);
        sX[tid * XSTRIDE + 65] = 0x3F80;          // 1.0bf16 -> multiplies bias row
    }

    // per-thread LSTM cell state: thread covers row m_e, cols nc..nc+15
    const int m_e  = tid >> 3;
    const int nc   = (tid & 7) * 16;
    const int fo_e = (tid & 7) * 8;
    float creg[16];
    #pragma unroll
    for (int j = 0; j < 16; ++j)
        creg[j] = c0[(size_t)(row0 + m_e) * HDIM + nc + j];
    const float bdv = bd[0];

    // elementwise-side base into the C-register-layout z buffer:
    // z(g, j) = zbase[g*2048 + j*8]
    const int m_lo = m_e & 15, m_hi = m_e >> 4;
    const float* zbase = &sZc[m_hi * 8192 + (tid & 7) * 256 +
                              ((m_lo >= 8) ? 128 : 0) + (m_lo & 7)];

    __syncthreads();

    const unsigned short* xrow_base = &sX[(mt * 16 + (lane & 15)) * XSTRIDE];
    const v8f vzero = {};

    // B tile loader: flat stage t -> (kt = t>>3, nt = t&7)
    auto loadB = [&](ABOp& dst, int t) {
        const int kt = t >> 3, nt = t & 7;
        const unsigned int* wp = &sW[(((kt << 5) + (g * 8 + nt)) * 32 + lane) * 8];
        dst.u4[0] = *(const uint4*)(wp);
        dst.u4[1] = *(const uint4*)(wp + 4);
    };
    auto loadA = [&](ABOp& dst, int kt) {
        const unsigned short* xp = xrow_base + kt * 32 + hlf * 16;
        dst.u4[0] = *(const uint4*)(xp);        // K 0..7   (+16*half)
        dst.u4[1] = *(const uint4*)(xp + 8);    // K 8..15  (+16*half)
    };

    // prologue: preload step-0 features into registers
    float4 fr0, fr1;
    {
        const float* fsrc = feats +
            ((size_t)(row0 + m_e) * TSTEPS + (TSTEPS - 1)) * FDIM + fo_e;
        fr0 = *(const float4*)(fsrc);
        fr1 = *(const float4*)(fsrc + 4);
    }

    for (int s = 0; s < TSTEPS; ++s) {
        // ---- commit this step's features (preloaded last iteration) as bf16 ----
        {
            unsigned int* dst = (unsigned int*)&sX[m_e * XSTRIDE + fo_e];
            dst[0] = packbf(fr0.x, fr0.y);
            dst[1] = packbf(fr0.z, fr0.w);
            dst[2] = packbf(fr1.x, fr1.y);
            dst[3] = packbf(fr1.z, fr1.w);
        }
        __syncthreads();   // orders feat + prev h/out writes before WMMA reads

        // ---- prefetch next step's features; latency hidden behind the GEMM ----
        if (s + 1 < TSTEPS) {
            const float* fsrc = feats +
                ((size_t)(row0 + m_e) * TSTEPS + (TSTEPS - 2 - s)) * FDIM + fo_e;
            fr0 = *(const float4*)(fsrc);
            fr1 = *(const float4*)(fsrc + 4);
        }

        // ---- GEMM: 16 rows x 128 gate cols, K=224 ----
        // 56 software-pipelined v_wmma_f32_16x16x32_bf16: B ring depth 4
        // (load t+2 ahead), A double-buffered (loaded mid-previous k-tile).
        v8f acc[8];
        #pragma unroll
        for (int nt = 0; nt < 8; ++nt) acc[nt] = vzero;

        ABOp a[2], bm[4];
        loadA(a[0], 0);
        loadB(bm[0], 0);
        loadB(bm[1], 1);
        #pragma unroll
        for (int t = 0; t < 56; ++t) {
            const int kt = t >> 3;
            const int nt = t & 7;
            if (t + 2 < 56) loadB(bm[(t + 2) & 3], t + 2);
            if (nt == 3 && kt < 6) loadA(a[(kt + 1) & 1], kt + 1);
            acc[nt] = __builtin_amdgcn_wmma_f32_16x16x32_bf16(
                false, a[kt & 1].v, false, bm[t & 3].v, (short)0, acc[nt],
                false, false);
        }

        // ---- spill z in native C layout: 2x ds_store_b128 per tile ----
        #pragma unroll
        for (int nt = 0; nt < 8; ++nt) {
            ACCU u; u.v = acc[nt];
            float* zp = &sZc[((wv * 8 + nt) * 32 + lane) * 8];
            *(float4*)(zp)     = u.f4[0];
            *(float4*)(zp + 4) = u.f4[1];
        }
        __syncthreads();

        // ---- elementwise LSTM cell + OUT projection (bias already in z) ----
        float hbuf[16];
        float dot = 0.0f;
        #pragma unroll
        for (int j = 0; j < 16; ++j) {
            const float zi = zbase[          j * 8];
            const float zf = zbase[2048 +    j * 8];
            const float zc = zbase[4096 +    j * 8];
            const float zo = zbase[6144 +    j * 8];
            const float cn = sigmoidf_(zf) * creg[j] + sigmoidf_(zi) * fast_tanh(zc);
            creg[j] = cn;
            const float hn = sigmoidf_(zo) * fast_tanh(cn);
            hbuf[j] = hn;
            dot += hn * sWd[nc + j];
        }
        {   // h -> bf16 back into x staging (K slots 66..193)
            unsigned int* hx = (unsigned int*)&sX[m_e * XSTRIDE + 66 + nc];
            #pragma unroll
            for (int j = 0; j < 8; ++j) hx[j] = packbf(hbuf[2 * j], hbuf[2 * j + 1]);
        }
        // reduce projection over the 8 threads covering this row (wave32 sub-groups)
        dot += __shfl_xor(dot, 1, 8);
        dot += __shfl_xor(dot, 2, 8);
        dot += __shfl_xor(dot, 4, 8);
        const float o = dot + bdv;
        if ((tid & 7) == 0) {
            out[(size_t)(row0 + m_e) * TSTEPS + s] = o;
            sX[m_e * XSTRIDE + 64] = f2bf(o);   // prev_out for next step
        }
        // next iteration's post-staging barrier orders these LDS writes
        // against the next step's WMMA reads
    }
}

extern "C" void kernel_launch(void* const* d_in, const int* in_sizes, int n_in,
                              void* d_out, int out_size, void* d_ws, size_t ws_size,
                              hipStream_t stream) {
    (void)in_sizes; (void)n_in; (void)out_size; (void)d_ws; (void)ws_size;
    const float* feats = (const float*)d_in[0];  // decoder_features [B,T,F]
    const float* dini  = (const float*)d_in[1];  // decoder_init_input [B,1]
    const float* h0    = (const float*)d_in[2];  // [B,H]
    const float* c0    = (const float*)d_in[3];  // [B,H]
    /* d_in[4] = encoder_output: unused by the reference computation */
    const float* Wx    = (const float*)d_in[5];  // [65,512]
    const float* Wh    = (const float*)d_in[6];  // [128,512]
    const float* b     = (const float*)d_in[7];  // [512]
    const float* Wd    = (const float*)d_in[8];  // [128,1]
    const float* bd    = (const float*)d_in[9];  // [1]
    float* out = (float*)d_out;                  // [B,T,1] f32

    dim3 grid(BATCH / ROWS);   // 256 workgroups
    dim3 block(256);           // 8 wave32 waves: (row-tile, gate)
    hipLaunchKernelGGL(lstm_decoder_wmma, grid, block, 0, stream,
                       feats, dini, h0, c0, Wx, Wh, b, Wd, bd, out);
}
